// V2VNetFusion_17394617549004
// MI455X (gfx1250) — matrix operand order repaired and checked by reading, not compiled, not run on hardware
//
#include <hip/hip_runtime.h>
#include <hip/hip_bf16.h>

typedef _Float16 f16;
typedef __attribute__((ext_vector_type(16))) _Float16 v16h;
typedef __attribute__((ext_vector_type(8)))  _Float16 v8h;
typedef __attribute__((ext_vector_type(8)))  float    v8f;

#define BB 2
#define LL 4
#define CC 128
#define HH 48
#define WW 176
#define NPIX (HH*WW)       // 8448
#define NIMG (BB*LL)       // 8
#define NPAIR (BB*LL*LL)   // 32
#define XTILES (WW/16)     // 11
#define NTILES (NPIX/16)   // 528
// zero-padded NHWC layout: 1-pixel halo on all sides
#define PW (WW+2)          // 178
#define PH (HH+2)          // 50
#define PPIX (PH*PW)       // 8900

// ---------------- zeroing (halo init; runs every launch, graph-safe) ----------------
__global__ void k_zero16(uint4* __restrict__ p, unsigned n16) {
  unsigned i = blockIdx.x*blockDim.x + threadIdx.x;
  if (i < n16) { uint4 z; z.x = z.y = z.z = z.w = 0u; p[i] = z; }
}

// ---------------- tiny setup kernels ----------------

// Apair[p] with p=((b*L+i)*L+j) holds the 2x3 inverse affine built from T[b,j,i]
__global__ void k_ainv(const float* __restrict__ T, float* __restrict__ Apair) {
  int p = threadIdx.x;
  if (p >= NPAIR) return;
  int b = p >> 4, i = (p >> 2) & 3, j = p & 3;
  const float* t = T + (((size_t)(b*LL + j)*LL + i) * 16);
  float R00 = t[0], R01 = t[1], R10 = t[4], R11 = t[5];
  float tx = t[3] / (0.4f * 4.0f), ty = t[7] / (0.4f * 4.0f);
  float cx = WW / 2.0f, cy = HH / 2.0f;
  float ftx = cx - (R00*cx + R01*cy) + tx;
  float fty = cy - (R10*cx + R11*cy) + ty;
  float det = R00*R11 - R01*R10;
  float i00 =  R11/det, i01 = -R01/det, i10 = -R10/det, i11 = R00/det;
  float tix = -(i00*ftx + i01*fty);
  float tiy = -(i10*ftx + i11*fty);
  float* a = Apair + p*6;
  a[0]=i00; a[1]=i01; a[2]=tix; a[3]=i10; a[4]=i11; a[5]=tiy;
}

// com_mask[p][pix] = warp(ones)[pix] * mask[b, j]   (unpadded, f32)
__global__ void k_commask(const float* __restrict__ Apair, const float* __restrict__ mask,
                          float* __restrict__ cm) {
  int tid = blockIdx.x*blockDim.x + threadIdx.x;
  if (tid >= NPAIR*NPIX) return;
  int p = tid / NPIX, pix = tid % NPIX;
  int y = pix / WW, x = pix % WW;
  const float* a = Apair + p*6;
  float sx = a[0]*x + a[1]*y + a[2];
  float sy = a[3]*x + a[4]*y + a[5];
  float x0 = floorf(sx), y0 = floorf(sy);
  float wx1 = sx - x0, wy1 = sy - y0;
  float roi = 0.f;
  #pragma unroll
  for (int cyi = 0; cyi < 2; ++cyi)
    #pragma unroll
    for (int cxi = 0; cxi < 2; ++cxi) {
      float xi = x0 + cxi, yi = y0 + cyi;
      float v = (xi >= 0.f && xi <= (float)(WW-1) && yi >= 0.f && yi <= (float)(HH-1)) ? 1.f : 0.f;
      float wgt = (cxi ? wx1 : 1.f - wx1) * (cyi ? wy1 : 1.f - wy1);
      roi += v * wgt;
    }
  int b = p >> 4, j = p & 3;
  cm[tid] = roi * mask[b*LL + j];
}

// x (f32 NCHW) -> F (f16 padded NHWC interior)
__global__ void k_xtof16(const float* __restrict__ x, f16* __restrict__ F) {
  size_t idx = (size_t)blockIdx.x*blockDim.x + threadIdx.x;
  if (idx >= (size_t)NIMG*CC*NPIX) return;
  int xx = (int)(idx % WW); size_t t = idx / WW;
  int y = (int)(t % HH); t /= HH;
  int c = (int)(t % CC); t /= CC;
  int img = (int)t;
  F[((size_t)img*PPIX + (size_t)(y+1)*PW + (xx+1))*CC + c] = (f16)x[idx];
}

// bilinear warp, padded-NHWC f16 in and out (interior only; halo stays zero)
__global__ void k_warp(const float* __restrict__ Apair, const f16* __restrict__ Fin,
                       f16* __restrict__ Wb) {
  int tid = blockIdx.x*blockDim.x + threadIdx.x;
  if (tid >= NPAIR*NPIX) return;
  int p = tid / NPIX, pix = tid % NPIX;
  int y = pix / WW, x = pix % WW;
  const float* a = Apair + p*6;
  float sx = a[0]*x + a[1]*y + a[2];
  float sy = a[3]*x + a[4]*y + a[5];
  float x0f = floorf(sx), y0f = floorf(sy);
  float wx1 = sx - x0f, wy1 = sy - y0f;
  int x0 = (int)x0f, y0 = (int)y0f;
  float wgt[4]; int off[4];
  #pragma unroll
  for (int cyi = 0; cyi < 2; ++cyi)
    #pragma unroll
    for (int cxi = 0; cxi < 2; ++cxi) {
      int k = cyi*2 + cxi;
      int xi = x0 + cxi, yi = y0 + cyi;
      bool v = (xi >= 0 && xi < WW && yi >= 0 && yi < HH);
      int xc = xi < 0 ? 0 : (xi > WW-1 ? WW-1 : xi);
      int yc = yi < 0 ? 0 : (yi > HH-1 ? HH-1 : yi);
      wgt[k] = (v ? 1.f : 0.f) * (cxi ? wx1 : 1.f - wx1) * (cyi ? wy1 : 1.f - wy1);
      off[k] = (yc+1)*PW + (xc+1);
    }
  int srcimg = (p >> 4)*LL + (p & 3);
  const f16* src = Fin + (size_t)srcimg*PPIX*CC;
  f16* dst = Wb + ((size_t)p*PPIX + (size_t)(y+1)*PW + (x+1))*CC;
  for (int c0 = 0; c0 < CC; c0 += 8) {
    v8h v00 = *(const v8h*)(src + (size_t)off[0]*CC + c0);
    v8h v01 = *(const v8h*)(src + (size_t)off[1]*CC + c0);
    v8h v10 = *(const v8h*)(src + (size_t)off[2]*CC + c0);
    v8h v11 = *(const v8h*)(src + (size_t)off[3]*CC + c0);
    v8h o;
    #pragma unroll
    for (int e = 0; e < 8; ++e)
      o[e] = (f16)(wgt[0]*(float)v00[e] + wgt[1]*(float)v01[e] +
                   wgt[2]*(float)v10[e] + wgt[3]*(float)v11[e]);
    *(v8h*)(dst + c0) = o;
  }
}

// ---- weight pre-pack, A layout with contiguous K order (k = 16*hi + s)
// so each lane reads its 16 A values as ONE contiguous 32B v16h at offset hi*16.
__device__ __forceinline__ int kperm(int kk) {
  return kk;
}

__global__ void k_pack_wmsg(const float* __restrict__ Wm, f16* __restrict__ P) {
  int idx = blockIdx.x*blockDim.x + threadIdx.x;
  if (idx >= 9*8*CC*32) return;
  int kk = idx & 31; int t = idx >> 5;
  int o = t % CC; int tk = t / CC;
  int kb = tk % 8; int tap = tk / 8;
  int ci = kb*32 + kk;
  int kh = tap/3, kw = tap%3;
  P[(size_t)t*32 + kperm(kk)] = (f16)Wm[(((size_t)o*(2*CC) + ci)*3 + kh)*3 + kw];
}

// rows 0..127 = u-gate (W_gates oc = C+m), rows 128..255 = candidate (W_can oc = m-C); ci in [0,2C)
__global__ void k_pack_wgru(const float* __restrict__ Wg, const float* __restrict__ Wc,
                            f16* __restrict__ P) {
  int idx = blockIdx.x*blockDim.x + threadIdx.x;
  if (idx >= 9*8*2*CC*32) return;
  int kk = idx & 31; int t = idx >> 5;
  int m = t % (2*CC); int tk = t / (2*CC);
  int kb = tk % 8; int tap = tk / 8;
  int ci = kb*32 + kk;
  int kh = tap/3, kw = tap%3;
  float v;
  if (m < CC) v = Wg[(((size_t)(CC + m)*(3*CC) + ci)*3 + kh)*3 + kw];
  else        v = Wc[(((size_t)(m - CC)*(3*CC) + ci)*3 + kh)*3 + kw];
  P[(size_t)t*32 + kperm(kk)] = (f16)v;
}

__global__ void k_pack_wmlp(const float* __restrict__ Wm, f16* __restrict__ P) {
  int idx = blockIdx.x*blockDim.x + threadIdx.x;
  if (idx >= 4*CC*32) return;
  int kk = idx & 31; int t = idx >> 5;
  int o = t % CC; int kb = t / CC;
  P[(size_t)t*32 + kperm(kk)] = (f16)Wm[(size_t)o*CC + kb*32 + kk];
}

// ---------------- message conv + mask + aggregate (WMMA implicit GEMM) ----------------
// block = 8 wave32s; wave w owns out-channels [16w,16w+16); block covers 32 pixels.
// ONE base pointer per N-subtile: neighbor j, tap and K-block are all folded into
// the 24-bit VGLOBAL immediate (max |offset| ~ 6.9 MB < 8.3 MB). Boundary taps read
// the zeroed halo -> no predication, no operand zeroing, no spills.
__global__ void k_msg(const f16* __restrict__ Wb, const f16* __restrict__ Fin,
                      const f16* __restrict__ WP, const float* __restrict__ cm,
                      const float* __restrict__ bmsg, const int* __restrict__ rl,
                      f16* __restrict__ Agg) {
  int img = blockIdx.y;
  int w = threadIdx.x >> 5, lane = threadIdx.x & 31;
  int lo = lane & 15, hi = lane >> 4;
  int chb = hi * 16;
  int nt0 = blockIdx.x * 2;
  int ys[2], xs[2], px[2];
  const f16* bWs[2];   // j==0 base; j folded into immediate
  const f16* bE[2];
  #pragma unroll
  for (int s = 0; s < 2; ++s) {
    int nt = nt0 + s;
    ys[s] = nt / XTILES;
    xs[s] = (nt % XTILES)*16 + lo;
    px[s] = ys[s]*WW + xs[s];
    size_t ppos = (size_t)(ys[s]+1)*PW + (xs[s]+1);
    bWs[s] = Wb + ((size_t)(img*LL)*PPIX + ppos)*CC + chb;
    bE[s]  = Fin + ((size_t)img*PPIX + ppos)*CC + chb;
  }
  const f16* aB = WP + (size_t)(16*w + lo)*32 + chb;  // chb == hi*16 chunk select

  v8f accW[LL][2] = {};   // per-neighbor (warped half, K=128)
  v8f accE[2]     = {};   // ego half (j-independent, K=128)

  #pragma unroll
  for (int tap = 0; tap < 9; ++tap) {
    const int doff = ((tap/3 - 1)*PW + (tap%3 - 1))*CC;   // compile-time constant
    #pragma unroll
    for (int kb = 0; kb < 8; ++kb) {
      v16h a = *(const v16h*)(aB + (tap*8 + kb)*(CC*32)); // constant offset
      if (kb < 4) {
        #pragma unroll
        for (int j = 0; j < LL; ++j) {
          const long joff = (long)j*(PPIX*CC) + doff + kb*32;  // compile-time constant
          #pragma unroll
          for (int s = 0; s < 2; ++s) {
            v16h bv = *(const v16h*)(bWs[s] + joff);
            accW[j][s] = __builtin_amdgcn_wmma_f32_16x16x32_f16(
                false, a, false, bv, (short)0, accW[j][s], false, false);
          }
        }
      } else {
        #pragma unroll
        for (int s = 0; s < 2; ++s) {
          v16h bv = *(const v16h*)(bE[s] + doff + (kb - 4)*32);
          accE[s] = __builtin_amdgcn_wmma_f32_16x16x32_f16(
              false, a, false, bv, (short)0, accE[s], false, false);
        }
      }
    }
  }
  float rinv = 1.f / (float)rl[img >> 2];
  float bias[8];
  #pragma unroll
  for (int e = 0; e < 8; ++e) bias[e] = bmsg[16*w + 8*hi + e];
  #pragma unroll
  for (int s = 0; s < 2; ++s) {
    float out[8];
    #pragma unroll
    for (int e = 0; e < 8; ++e) out[e] = 0.f;
    #pragma unroll
    for (int j = 0; j < LL; ++j) {
      float m = cm[(size_t)(img*LL + j)*NPIX + px[s]];
      #pragma unroll
      for (int e = 0; e < 8; ++e)
        out[e] += (accW[j][s][e] + accE[s][e] + bias[e]) * m;   // bias before mask, per reference
    }
    v8h o;
    #pragma unroll
    for (int e = 0; e < 8; ++e) o[e] = (f16)(out[e] * rinv);
    *(v8h*)(Agg + ((size_t)img*PPIX + (size_t)(ys[s]+1)*PW + (xs[s]+1))*CC + 16*w + 8*hi) = o;
  }
}

// ---------------- fused GRU: h = sigmoid(conv_u(in)) * tanh(conv_c(in)) ----------------
__global__ void k_gru(const f16* __restrict__ Fin, const f16* __restrict__ Agg,
                      const f16* __restrict__ WP, const float* __restrict__ bg,
                      const float* __restrict__ bc, f16* __restrict__ Fout) {
  int img = blockIdx.y;
  int w = threadIdx.x >> 5, lane = threadIdx.x & 31;
  int lo = lane & 15, hi = lane >> 4;
  int chb = hi * 16;
  int nt0 = blockIdx.x * 4;
  int ys[4], xs[4];
  const f16* bF[4];
  const f16* bA[4];
  #pragma unroll
  for (int s = 0; s < 4; ++s) {
    int nt = nt0 + s;
    ys[s] = nt / XTILES; xs[s] = (nt % XTILES)*16 + lo;
    size_t ppos = (size_t)(ys[s]+1)*PW + (xs[s]+1);
    bF[s] = Fin + ((size_t)img*PPIX + ppos)*CC + chb;
    bA[s] = Agg + ((size_t)img*PPIX + ppos)*CC + chb;
  }
  const f16* aB = WP + (size_t)(16*w + lo)*32 + chb;

  v8f accU[4] = {}, accC[4] = {};
  #pragma unroll
  for (int tap = 0; tap < 9; ++tap) {
    const int doff = ((tap/3 - 1)*PW + (tap%3 - 1))*CC;
    #pragma unroll
    for (int kb = 0; kb < 8; ++kb) {
      v16h a1 = *(const v16h*)(aB + (tap*8 + kb)*(2*CC*32));           // u rows
      v16h a2 = *(const v16h*)(aB + (tap*8 + kb)*(2*CC*32) + CC*32);   // cand rows
      #pragma unroll
      for (int s = 0; s < 4; ++s) {
        v16h bv;
        if (kb < 4) bv = *(const v16h*)(bF[s] + doff + kb*32);
        else        bv = *(const v16h*)(bA[s] + doff + (kb - 4)*32);
        accU[s] = __builtin_amdgcn_wmma_f32_16x16x32_f16(
            false, a1, false, bv, (short)0, accU[s], false, false);
        accC[s] = __builtin_amdgcn_wmma_f32_16x16x32_f16(
            false, a2, false, bv, (short)0, accC[s], false, false);
      }
    }
  }
  #pragma unroll
  for (int s = 0; s < 4; ++s) {
    v8h o;
    #pragma unroll
    for (int e = 0; e < 8; ++e) {
      int oe = 16*w + 8*hi + e;
      float u = 1.f / (1.f + __expf(-(accU[s][e] + bg[CC + oe])));
      float c = tanhf(accC[s][e] + bc[oe]);
      o[e] = (f16)(u * c);
    }
    *(v8h*)(Fout + ((size_t)img*PPIX + (size_t)(ys[s]+1)*PW + (xs[s]+1))*CC + 16*w + 8*hi) = o;
  }
}

// ---------------- final 1x1 MLP (WMMA GEMM), writes f32 NCHW to d_out ----------------
__global__ void k_mlp(const f16* __restrict__ Fin, const f16* __restrict__ WP,
                      const float* __restrict__ bm, float* __restrict__ out) {
  int b = blockIdx.y;
  int img = b * LL;       // feat[:, 0]
  int w = threadIdx.x >> 5, lane = threadIdx.x & 31;
  int lo = lane & 15, hi = lane >> 4;
  int chb = hi * 16;
  int nt = blockIdx.x;
  int y = nt / XTILES, x = (nt % XTILES)*16 + lo;
  int px = y*WW + x;
  const f16* bB = Fin + ((size_t)img*PPIX + (size_t)(y+1)*PW + (x+1))*CC + chb;
  const f16* aB = WP + (size_t)(16*w + lo)*32 + chb;
  v8f acc = {};
  #pragma unroll
  for (int kb = 0; kb < 4; ++kb) {
    v16h a  = *(const v16h*)(aB + kb*(CC*32));
    v16h bv = *(const v16h*)(bB + kb*32);
    acc = __builtin_amdgcn_wmma_f32_16x16x32_f16(false, a, false, bv, (short)0, acc, false, false);
  }
  #pragma unroll
  for (int e = 0; e < 8; ++e) {
    int oe = 16*w + 8*hi + e;
    out[((size_t)b*CC + oe)*NPIX + px] = acc[e] + bm[oe];
  }
}

// ---------------- host side ----------------
extern "C" void kernel_launch(void* const* d_in, const int* in_sizes, int n_in,
                              void* d_out, int out_size, void* d_ws, size_t ws_size,
                              hipStream_t stream) {
  const float* x    = (const float*)d_in[0];
  const float* mask = (const float*)d_in[1];
  const float* pT   = (const float*)d_in[2];
  const int*   rl   = (const int*)d_in[3];
  const float* Wmsg = (const float*)d_in[4];
  const float* bmsg = (const float*)d_in[5];
  const float* Wg   = (const float*)d_in[6];
  const float* bg   = (const float*)d_in[7];
  const float* Wc   = (const float*)d_in[8];
  const float* bc   = (const float*)d_in[9];
  const float* Wmlp = (const float*)d_in[10];
  const float* bmlp = (const float*)d_in[11];
  (void)in_sizes; (void)n_in; (void)out_size; (void)ws_size;

  char* ws = (char*)d_ws;
  size_t off = 0;
  auto alloc = [&](size_t bytes) -> void* {
    void* p = ws + off;
    off += (bytes + 255) & ~(size_t)255;
    return p;
  };
  float* Apair = (float*)alloc((size_t)NPAIR*6*sizeof(float));
  float* cmask = (float*)alloc((size_t)NPAIR*NPIX*sizeof(float));
  f16* WmsgP = (f16*)alloc((size_t)9*8*CC*32*sizeof(f16));
  f16* WgruP = (f16*)alloc((size_t)9*8*2*CC*32*sizeof(f16));
  f16* WmlpP = (f16*)alloc((size_t)4*CC*32*sizeof(f16));
  // the four padded f16 activation buffers are contiguous -> one zeroing pass
  size_t fbytes  = (size_t)NIMG*PPIX*CC*sizeof(f16);   // 18,227,200 (256-aligned)
  size_t wbytes  = (size_t)NPAIR*PPIX*CC*sizeof(f16);  // 72,908,800
  char* zbase = (char*)alloc(fbytes*3 + wbytes);
  f16* F0   = (f16*)(zbase);
  f16* F1   = (f16*)(zbase + fbytes);
  f16* AggB = (f16*)(zbase + 2*fbytes);
  f16* Wbuf = (f16*)(zbase + 3*fbytes);

  unsigned n16 = (unsigned)((fbytes*3 + wbytes) / 16);
  k_zero16<<<(n16 + 255)/256, 256, 0, stream>>>((uint4*)zbase, n16);

  k_ainv<<<1, 32, 0, stream>>>(pT, Apair);
  k_commask<<<(NPAIR*NPIX + 255)/256, 256, 0, stream>>>(Apair, mask, cmask);
  {
    size_t n = (size_t)NIMG*CC*NPIX;
    k_xtof16<<<(unsigned)((n + 255)/256), 256, 0, stream>>>(x, F0);
  }
  k_pack_wmsg<<<(9*8*CC*32 + 255)/256, 256, 0, stream>>>(Wmsg, WmsgP);
  k_pack_wgru<<<(9*8*2*CC*32 + 255)/256, 256, 0, stream>>>(Wg, Wc, WgruP);
  k_pack_wmlp<<<(4*CC*32 + 255)/256, 256, 0, stream>>>(Wmlp, WmlpP);

  f16* cur = F0; f16* nxt = F1;
  for (int it = 0; it < 2; ++it) {
    k_warp<<<(NPAIR*NPIX + 255)/256, 256, 0, stream>>>(Apair, cur, Wbuf);
    k_msg<<<dim3(NTILES/2, NIMG), 256, 0, stream>>>(Wbuf, cur, WmsgP, cmask, bmsg, rl, AggB);
    k_gru<<<dim3(NTILES/4, NIMG), 256, 0, stream>>>(cur, AggB, WgruP, bg, bc, nxt);
    f16* t = cur; cur = nxt; nxt = t;
  }
  k_mlp<<<dim3(NTILES, BB), 256, 0, stream>>>(cur, WmlpP, bmlp, (float*)d_out);
}
